// PairRankNet_21191368639244
// MI455X (gfx1250) — compile-verified, matrix operand order checked by reading, not run on hardware
//
#include <hip/hip_runtime.h>
#include <hip/hip_bf16.h>
#include <math.h>

typedef _Float16 f16;
typedef _Float16 v8h  __attribute__((ext_vector_type(8)));
typedef _Float16 v16h __attribute__((ext_vector_type(16)));
typedef float    v8f  __attribute__((ext_vector_type(8)));

#define B_     64
#define M_     5
#define N_     16384
#define NB_    128
#define D_     128
#define NP_    10
#define PAIRS_ 8128   // 128*127/2

__constant__ int   c_mi[NP_]   = {0,0,0,0,1,1,1,2,2,3};
__constant__ int   c_mj[NP_]   = {1,2,3,4,2,3,4,3,4,4};
__constant__ float c_micp[15]  = {0.f,0.f,1000.f, 3000.f,0.f,1000.f, 0.f,3000.f,1000.f,
                                  3000.f,3000.f,1000.f, 1500.f,1500.f,2000.f};

// ---------------------------------------------------------------------------
// 1) Forward FFT per (b,mic): normalize (ddof=1), DIF radix-2 in LDS.
//    Output spectrum is bit-reversed-order; consistent for per-bin PHAT.
//    128 KB complex buffer lives entirely in CDNA5's 320 KB/WGP LDS.
// ---------------------------------------------------------------------------
__global__ void fft_fwd_kernel(const float* __restrict__ audio, float2* __restrict__ spec) {
  extern __shared__ float2 buf[];              // N_ complex fp32 = 128 KB in LDS
  const int tid = threadIdx.x;                 // 512 threads
  const float* x = audio + (size_t)blockIdx.x * N_;

  float s = 0.f, s2 = 0.f;
  for (int i = tid; i < N_; i += 512) { float v = x[i]; s += v; s2 += v * v; }
  buf[tid] = make_float2(s, s2);
  __syncthreads();
  for (int off = 256; off > 0; off >>= 1) {
    if (tid < off) { buf[tid].x += buf[tid + off].x; buf[tid].y += buf[tid + off].y; }
    __syncthreads();
  }
  const float mean = buf[0].x * (1.0f / N_);
  const float var  = (buf[0].y - (float)N_ * mean * mean) * (1.0f / (N_ - 1));
  const float inv  = 1.0f / (sqrtf(fmaxf(var, 0.f)) + 1e-8f);
  __syncthreads();

  for (int i = tid; i < N_; i += 512) buf[i] = make_float2((x[i] - mean) * inv, 0.f);
  __syncthreads();

  const float twoPi = 6.283185307179586f;
  for (int st = 13; st >= 0; --st) {           // DIF: half = 8192 .. 1
    const int half = 1 << st;
    for (int t = tid; t < (N_ / 2); t += 512) {
      const int jj   = t & (half - 1);
      const int base = (t >> st) << (st + 1);
      const int i0 = base + jj, i1 = i0 + half;
      float2 a = buf[i0], b = buf[i1];
      float2 d = make_float2(a.x - b.x, a.y - b.y);
      float sn, cs;
      __sincosf(-twoPi * (float)jj / (float)(half << 1), &sn, &cs);
      buf[i0] = make_float2(a.x + b.x, a.y + b.y);
      buf[i1] = make_float2(d.x * cs - d.y * sn, d.x * sn + d.y * cs);
    }
    __syncthreads();
  }
  float2* out = spec + (size_t)blockIdx.x * N_;
  for (int i = tid; i < N_; i += 512) out[i] = buf[i];
}

// ---------------------------------------------------------------------------
// 2) Per (b,pair): cross-spectrum, PHAT normalize, inverse DIT FFT in LDS,
//    then peak / mean / sharp stats (fftshift-concat duplicates bin n/2).
// ---------------------------------------------------------------------------
__global__ void gcc_phat_kernel(const float2* __restrict__ spec, float* __restrict__ stats) {
  extern __shared__ float2 buf[];              // N_ + 512 float2
  float2* red = buf + N_;
  const int tid = threadIdx.x;
  const int b = blockIdx.x / NP_, p = blockIdx.x % NP_;
  const float2* Si = spec + (size_t)(b * M_ + c_mi[p]) * N_;
  const float2* Sj = spec + (size_t)(b * M_ + c_mj[p]) * N_;

  for (int i = tid; i < N_; i += 512) {
    float2 a = Si[i], c = Sj[i];
    float cx = a.x * c.x + a.y * c.y;          // a * conj(c)
    float cy = a.y * c.x - a.x * c.y;
    float mg = sqrtf(cx * cx + cy * cy) + 1e-10f;
    buf[i] = make_float2(cx / mg, cy / mg);
  }
  __syncthreads();

  const float twoPi = 6.283185307179586f;
  for (int st = 0; st <= 13; ++st) {           // DIT inverse: bit-rev in -> natural out
    const int half = 1 << st;
    for (int t = tid; t < (N_ / 2); t += 512) {
      const int jj   = t & (half - 1);
      const int base = (t >> st) << (st + 1);
      const int i0 = base + jj, i1 = i0 + half;
      float sn, cs;
      __sincosf(twoPi * (float)jj / (float)(half << 1), &sn, &cs);
      float2 bb = buf[i1];
      float2 tw = make_float2(bb.x * cs - bb.y * sn, bb.x * sn + bb.y * cs);
      float2 a  = buf[i0];
      buf[i0] = make_float2(a.x + tw.x, a.y + tw.y);
      buf[i1] = make_float2(a.x - tw.x, a.y - tw.y);
    }
    __syncthreads();
  }

  const float invN = 1.0f / (float)N_;
  float pk = -INFINITY, sm = 0.f;
  for (int i = tid; i < N_; i += 512) {
    float v = buf[i].x * invN;
    pk = fmaxf(pk, v); sm += v;
  }
  red[tid] = make_float2(pk, sm);
  __syncthreads();
  for (int off = 256; off > 0; off >>= 1) {
    if (tid < off) {
      red[tid].x = fmaxf(red[tid].x, red[tid + off].x);
      red[tid].y += red[tid + off].y;
    }
    __syncthreads();
  }
  if (tid == 0) {
    const float peak  = red[0].x;
    const float mean  = (red[0].y + buf[N_ / 2].x * invN) / (float)(N_ + 1);
    const float sharp = peak / (fabsf(mean) + 1e-6f);
    float* o = stats + (size_t)b * 30 + p * 3;
    o[0] = peak; o[1] = mean; o[2] = sharp;
  }
}

// ---------------------------------------------------------------------------
// 3) Tiny audio MLP 30 -> 128 -> 128 (fp32, one block per batch)
// ---------------------------------------------------------------------------
__global__ void audio_mlp_kernel(const float* __restrict__ stats,
                                 const float* __restrict__ Wa1, const float* __restrict__ ba1,
                                 const float* __restrict__ Wa2, const float* __restrict__ ba2,
                                 float* __restrict__ aemb) {
  __shared__ float h[D_];
  const int b = blockIdx.x, t = threadIdx.x;
  const float* st = stats + (size_t)b * 30;
  float acc = ba1[t];
  for (int k = 0; k < 30; ++k) acc += st[k] * Wa1[k * D_ + t];
  h[t] = fmaxf(acc, 0.f);
  __syncthreads();
  float o = ba2[t];
  for (int k = 0; k < D_; ++k) o += h[k] * Wa2[k * D_ + t];
  aemb[(size_t)b * D_ + t] = o;
}

// ---------------------------------------------------------------------------
// 4) Weight prep: f16 transposed [N,K] layouts for the WMMA B-operand.
// ---------------------------------------------------------------------------
__global__ void prep_weights_kernel(const float* __restrict__ Wc1, const float* __restrict__ Wc2,
                                    const float* __restrict__ Wp1,
                                    f16* __restrict__ Wc1t, f16* __restrict__ Wc2t,
                                    f16* __restrict__ Wp1at, f16* __restrict__ Wp1bt) {
  const int idx = blockIdx.x * blockDim.x + threadIdx.x;
  if (idx < 4096) {                                  // Wc1t [128][32], K padded 28->32
    const int n = idx >> 5, k = idx & 31;
    Wc1t[idx] = (k < 28) ? (f16)Wc1[k * D_ + n] : (f16)0.f;
  } else if (idx < 4096 + 16384) {
    const int r = idx - 4096, n = r >> 7, k = r & 127;
    Wc2t[r] = (f16)Wc2[k * D_ + n];
  } else if (idx < 4096 + 32768) {
    const int r = idx - 4096 - 16384, n = r >> 7, k = r & 127;
    Wp1at[r] = (f16)Wp1[k * D_ + n];                 // Wp1 rows 0..127
  } else if (idx < 4096 + 49152) {
    const int r = idx - 4096 - 32768, n = r >> 7, k = r & 127;
    Wp1bt[r] = (f16)Wp1[(128 + k) * D_ + n];         // Wp1 rows 128..255
  }
}

// ---------------------------------------------------------------------------
// 5) Candidate features: [pos(3) head(3) radio(12) expected(10) pad(4)] f16,
//    plus expected-TDOA kept in fp32 for the pair stage.
// ---------------------------------------------------------------------------
__global__ void feat_kernel(const float* __restrict__ pos, const float* __restrict__ head,
                            const float* __restrict__ radio,
                            f16* __restrict__ feat, float* __restrict__ expct) {
  const int idx = blockIdx.x * blockDim.x + threadIdx.x;
  if (idx >= B_ * NB_) return;
  const float p0 = pos[idx * 3], p1 = pos[idx * 3 + 1], p2 = pos[idx * 3 + 2];
  const float x = p0 * 3000.f, y = p1 * 3000.f, z = p2 * 2500.f;
  float d[M_];
#pragma unroll
  for (int k = 0; k < M_; ++k) {
    const float dx = x - c_micp[k * 3], dy = y - c_micp[k * 3 + 1], dz = z - c_micp[k * 3 + 2];
    d[k] = sqrtf(dx * dx + dy * dy + dz * dz);
  }
  f16* f = feat + (size_t)idx * 32;
  f[0] = (f16)p0; f[1] = (f16)p1; f[2] = (f16)p2;
#pragma unroll
  for (int k = 0; k < 3; ++k)  f[3 + k] = (f16)head[idx * 3 + k];
#pragma unroll
  for (int k = 0; k < 12; ++k) f[6 + k] = (f16)radio[idx * 12 + k];
  const float kconv = 24414.0625f / 343000.0f / 50.0f;
#pragma unroll
  for (int p = 0; p < NP_; ++p) {
    const float e = (d[c_mi[p]] - d[c_mj[p]]) * kconv;
    expct[(size_t)idx * NP_ + p] = e;
    f[18 + p] = (f16)e;
  }
  f[28] = f[29] = f[30] = f[31] = (f16)0.f;
}

// ---------------------------------------------------------------------------
// 6) WMMA GEMM: C[M,128] = A[M,K]f16 @ Wt[128,K]^T f16 (+bias, opt relu).
//    One wave per 16x16 tile; 8 waves/block cover all 128 columns of 16 rows.
//    A/B per-lane layout per CDNA5 ISA 7.12.2: lanes 0-15 hold K {0..7,16..23},
//    lanes 16-31 hold K {8..15,24..31} -> two contiguous v8h (16B) loads each.
//    All epilogue options are template params -> straight-line store path
//    (lanes 0-15 / 16-31 each cover 16 consecutive columns: coalesced b32/b16).
// ---------------------------------------------------------------------------
template <int K, bool RELU, bool HAS_BIAS, bool OUT_F32, bool OUT_F16>
__global__ void gemm_wmma_kernel(const f16* __restrict__ A, const f16* __restrict__ Wt,
                                 const float* __restrict__ bias,
                                 float* __restrict__ Cf, f16* __restrict__ Ch, int Mrows) {
  const int lane = threadIdx.x & 31;
  const int wave = threadIdx.x >> 5;
  const int row0 = blockIdx.x << 4;
  const int col0 = wave << 4;
  if (row0 >= Mrows) return;
  const int mn  = lane & 15;
  const int grp = lane >> 4;

  const f16* pa0 = A  + (size_t)(row0 + mn) * K + grp * 8;
  const f16* pb0 = Wt + (size_t)(col0 + mn) * K + grp * 8;

  v8f acc = {};
#pragma unroll
  for (int kk = 0; kk < K; kk += 32) {
    v8h alo = *(const v8h*)(pa0 + kk);
    v8h ahi = *(const v8h*)(pa0 + kk + 16);
    v8h blo = *(const v8h*)(pb0 + kk);
    v8h bhi = *(const v8h*)(pb0 + kk + 16);
    v16h av, bv;
#pragma unroll
    for (int t = 0; t < 8; ++t) {
      av[t] = alo[t]; av[t + 8] = ahi[t];
      bv[t] = blo[t]; bv[t + 8] = bhi[t];
    }
    acc = __builtin_amdgcn_wmma_f32_16x16x32_f16(
        /*neg_a=*/false, av, /*neg_b=*/false, bv,
        /*c_mod=*/(short)0, acc, /*reuse_a=*/false, /*reuse_b=*/false);
  }

  const float bn = HAS_BIAS ? bias[col0 + mn] : 0.0f;
#pragma unroll
  for (int r = 0; r < 8; ++r) {                 // C/D layout: lanes 0-15 rows 0-7, 16-31 rows 8-15
    float v = acc[r] + bn;
    if (RELU) v = fmaxf(v, 0.0f);
    const size_t off = (size_t)(row0 + r + grp * 8) * D_ + col0 + mn;
    if (OUT_F32) Cf[off] = v;
    if (OUT_F16) Ch[off] = (f16)v;
  }
}

// ---------------------------------------------------------------------------
// 7) Pair scorer: one wave per (b, pair). pre = U_i + V_j + geo@Wp1_geo + b,
//    pref = relu(pre) @ Wp2 + bp2 -> deterministic per-pair buffer.
// ---------------------------------------------------------------------------
__global__ void pair_score_kernel(const float* __restrict__ U, const float* __restrict__ V,
                                  const float* __restrict__ pos, const float* __restrict__ expct,
                                  const float* __restrict__ Wp1, const float* __restrict__ bp1,
                                  const float* __restrict__ Wp2, const float* __restrict__ bp2,
                                  float* __restrict__ pref_out) {
  const int w = (blockIdx.x << 3) + (threadIdx.x >> 5);
  if (w >= B_ * PAIRS_) return;
  const int lane = threadIdx.x & 31;
  const int b = w / PAIRS_;
  const int q = w - b * PAIRS_;

  // invert triangular index: i = largest row with f(i) <= q, f(i)=i*(255-i)/2
  int i = (int)floorf((255.0f - sqrtf(255.0f * 255.0f - 8.0f * (float)q)) * 0.5f);
  i = i < 0 ? 0 : (i > 126 ? 126 : i);
  while (i < 126 && ((i + 1) * (255 - (i + 1))) / 2 <= q) ++i;
  while (i > 0 && (i * (255 - i)) / 2 > q) --i;
  const int j = q - (i * (255 - i)) / 2 + i + 1;

  const int gi = b * NB_ + i, gj = b * NB_ + j;
  const float rx = pos[gi * 3 + 0] - pos[gj * 3 + 0];
  const float ry = pos[gi * 3 + 1] - pos[gj * 3 + 1];
  const float rz = pos[gi * 3 + 2] - pos[gj * 3 + 2];
  const float rd = sqrtf(rx * rx + ry * ry + rz * rz);
  float td = 0.f;
#pragma unroll
  for (int p = 0; p < NP_; ++p) td += expct[(size_t)gi * NP_ + p] - expct[(size_t)gj * NP_ + p];
  td *= 0.1f;
  const float geo[5] = {rx, ry, rz, rd, td};

  float part = 0.0f;
#pragma unroll
  for (int rep = 0; rep < 4; ++rep) {
    const int t = lane + rep * 32;
    float pre = U[(size_t)gi * D_ + t] + V[(size_t)gj * D_ + t] + bp1[t];
#pragma unroll
    for (int g = 0; g < 5; ++g) pre += geo[g] * Wp1[(256 + g) * D_ + t];
    part += fmaxf(pre, 0.0f) * Wp2[t];
  }
#pragma unroll
  for (int off = 16; off > 0; off >>= 1) part += __shfl_xor(part, off, 32);
  if (lane == 0) pref_out[(size_t)b * PAIRS_ + q] = part + bp2[0];
}

// 8) Gather per-candidate score sums (deterministic, no atomics).
__global__ void accum_scores_kernel(const float* __restrict__ pref, float* __restrict__ scores) {
  const int idx = blockIdx.x * blockDim.x + threadIdx.x;
  if (idx >= B_ * NB_) return;
  const int b = idx >> 7, i = idx & 127;
  const float* pp = pref + (size_t)b * PAIRS_;
  float s = 0.f;
  for (int j = 0; j < NB_; ++j) {
    if (j == i) continue;
    if (j > i) s += pp[(i * (255 - i)) / 2 + (j - i - 1)];
    else       s -= pp[(j * (255 - j)) / 2 + (i - j - 1)];
  }
  scores[idx] = s;
}

// 9) Finalize: += 0.05*<cand_emb, audio_emb>, mask, softmax, pred_pos.
__global__ void finalize_kernel(const float* __restrict__ scores_in,
                                const float* __restrict__ cand32,
                                const float* __restrict__ aemb,
                                const float* __restrict__ pos,
                                const unsigned char* __restrict__ mask,
                                float* __restrict__ out) {
  __shared__ float sh[NB_];
  const int b = blockIdx.x, c = threadIdx.x;
  const float* ce = cand32 + ((size_t)b * NB_ + c) * D_;
  const float* ae = aemb + (size_t)b * D_;
  float dot = 0.f;
  for (int t = 0; t < D_; ++t) dot += ce[t] * ae[t];
  float s = scores_in[b * NB_ + c] + 0.05f * dot;
  if (!mask[b * NB_ + c]) s = -INFINITY;
  out[b * NB_ + c] = s;

  sh[c] = s; __syncthreads();
  for (int off = 64; off > 0; off >>= 1) { if (c < off) sh[c] = fmaxf(sh[c], sh[c + off]); __syncthreads(); }
  const float mx = sh[0]; __syncthreads();
  const float e = __expf(s - mx);
  sh[c] = e; __syncthreads();
  for (int off = 64; off > 0; off >>= 1) { if (c < off) sh[c] += sh[c + off]; __syncthreads(); }
  const float attn = e / sh[0]; __syncthreads();

  for (int d = 0; d < 3; ++d) {
    sh[c] = attn * pos[((size_t)b * NB_ + c) * 3 + d]; __syncthreads();
    for (int off = 64; off > 0; off >>= 1) { if (c < off) sh[c] += sh[c + off]; __syncthreads(); }
    if (c == 0) out[B_ * NB_ + b * 3 + d] = sh[0];
    __syncthreads();
  }
}

// ---------------------------------------------------------------------------
extern "C" void kernel_launch(void* const* d_in, const int* in_sizes, int n_in,
                              void* d_out, int out_size, void* d_ws, size_t ws_size,
                              hipStream_t stream) {
  (void)in_sizes; (void)n_in; (void)out_size; (void)ws_size;
  const float* audio = (const float*)d_in[0];
  const float* pos   = (const float*)d_in[1];
  const float* head  = (const float*)d_in[2];
  const float* radio = (const float*)d_in[3];
  const unsigned char* mask = (const unsigned char*)d_in[4];
  const float* Wa1 = (const float*)d_in[5];
  const float* ba1 = (const float*)d_in[6];
  const float* Wa2 = (const float*)d_in[7];
  const float* ba2 = (const float*)d_in[8];
  const float* Wc1 = (const float*)d_in[9];
  const float* bc1 = (const float*)d_in[10];
  const float* Wc2 = (const float*)d_in[11];
  const float* bc2 = (const float*)d_in[12];
  const float* Wp1 = (const float*)d_in[13];
  const float* bp1 = (const float*)d_in[14];
  const float* Wp2 = (const float*)d_in[15];
  const float* bp2 = (const float*)d_in[16];
  float* out = (float*)d_out;

  char* w = (char*)d_ws;
  auto alloc = [&](size_t bytes) -> char* {
    char* p = w; w += (bytes + 255) & ~(size_t)255; return p;
  };
  float2* spec   = (float2*)alloc((size_t)B_ * M_ * N_ * sizeof(float2));   // 40 MB
  float*  stats  = (float*) alloc((size_t)B_ * 30 * sizeof(float));
  float*  aemb   = (float*) alloc((size_t)B_ * D_ * sizeof(float));
  f16*    feat16 = (f16*)   alloc((size_t)B_ * NB_ * 32 * sizeof(f16));
  float*  expct  = (float*) alloc((size_t)B_ * NB_ * NP_ * sizeof(float));
  f16*    h1     = (f16*)   alloc((size_t)B_ * NB_ * D_ * sizeof(f16));
  f16*    cand16 = (f16*)   alloc((size_t)B_ * NB_ * D_ * sizeof(f16));
  float*  cand32 = (float*) alloc((size_t)B_ * NB_ * D_ * sizeof(float));
  float*  Uf     = (float*) alloc((size_t)B_ * NB_ * D_ * sizeof(float));
  float*  Vf     = (float*) alloc((size_t)B_ * NB_ * D_ * sizeof(float));
  float*  prefb  = (float*) alloc((size_t)B_ * PAIRS_ * sizeof(float));
  float*  scores = (float*) alloc((size_t)B_ * NB_ * sizeof(float));
  f16*    Wc1t   = (f16*)   alloc((size_t)128 * 32 * sizeof(f16));
  f16*    Wc2t   = (f16*)   alloc((size_t)128 * 128 * sizeof(f16));
  f16*    Wp1at  = (f16*)   alloc((size_t)128 * 128 * sizeof(f16));
  f16*    Wp1bt  = (f16*)   alloc((size_t)128 * 128 * sizeof(f16));

  const int rowsM = B_ * NB_;                   // 8192

  // Audio branch
  fft_fwd_kernel<<<B_ * M_, 512, N_ * sizeof(float2), stream>>>(audio, spec);
  gcc_phat_kernel<<<B_ * NP_, 512, (N_ + 512) * sizeof(float2), stream>>>(spec, stats);
  audio_mlp_kernel<<<B_, D_, 0, stream>>>(stats, Wa1, ba1, Wa2, ba2, aemb);

  // Candidate branch
  prep_weights_kernel<<<(4096 + 49152 + 255) / 256, 256, 0, stream>>>(
      Wc1, Wc2, Wp1, Wc1t, Wc2t, Wp1at, Wp1bt);
  feat_kernel<<<(rowsM + 255) / 256, 256, 0, stream>>>(pos, head, radio, feat16, expct);

  // layer1 + relu -> h1 (f16 only)
  gemm_wmma_kernel<32, true, true, false, true><<<rowsM / 16, 256, 0, stream>>>(
      feat16, Wc1t, bc1, nullptr, h1, rowsM);
  // layer2 -> cand_emb (f32 + f16)
  gemm_wmma_kernel<128, false, true, true, true><<<rowsM / 16, 256, 0, stream>>>(
      h1, Wc2t, bc2, cand32, cand16, rowsM);
  // U = emb @ Wp1_top (f32 only, no bias)
  gemm_wmma_kernel<128, false, false, true, false><<<rowsM / 16, 256, 0, stream>>>(
      cand16, Wp1at, nullptr, Uf, nullptr, rowsM);
  // V = emb @ Wp1_bot (f32 only, no bias)
  gemm_wmma_kernel<128, false, false, true, false><<<rowsM / 16, 256, 0, stream>>>(
      cand16, Wp1bt, nullptr, Vf, nullptr, rowsM);

  // Pair scoring + outputs
  pair_score_kernel<<<(B_ * PAIRS_) / 8, 256, 0, stream>>>(
      Uf, Vf, pos, expct, Wp1, bp1, Wp2, bp2, prefb);
  accum_scores_kernel<<<(rowsM + 255) / 256, 256, 0, stream>>>(prefb, scores);
  finalize_kernel<<<B_, NB_, 0, stream>>>(scores, cand32, aemb, pos, mask, out);
}
